// EdgeBlockCUGOSum_87162066305236
// MI455X (gfx1250) — compile-verified, matrix operand order checked by reading, not run on hardware
//
#include <hip/hip_runtime.h>
#include <hip/hip_bf16.h>

typedef __attribute__((ext_vector_type(16))) _Float16 v16h;
typedef __attribute__((ext_vector_type(8)))  _Float16 v8h;
typedef __attribute__((ext_vector_type(4)))  _Float16 v4h;
typedef __attribute__((ext_vector_type(8)))  float    v8f;
typedef __attribute__((ext_vector_type(4)))  float    v4f;

#define EDGES 600000
#define NNODES 100000
#define DIM 128
#define LDSW 136              // padded f16 row stride (conflict-free 16B lane reads)
#define YSTR 132              // padded f32 row stride for LN staging

// ---------------------------------------------------------------------------
// Prep: W1 (384x128 f32) and W2 (128x128 f32) -> f16, transposed to [n][k]
// layout in d_ws: [w][n][k], w in {Wf, Ws, Wd, W2}, each 128x128 halfs.
// ---------------------------------------------------------------------------
extern "C" __global__ __launch_bounds__(256)
void prep_weights_f16(const float* __restrict__ W1, const float* __restrict__ W2,
                      _Float16* __restrict__ w16) {
  int g = blockIdx.x * 256 + threadIdx.x;     // 0..16383 (groups of 4 k)
  int w   = g >> 12;                          // matrix 0..3
  int rem = g & 4095;
  int n   = rem >> 5;                         // output column
  int kg  = (rem & 31) * 4;                   // k group base
  v4h hv;
  if (w < 3) {
#pragma unroll
    for (int i = 0; i < 4; ++i) hv[i] = (_Float16)W1[(w * 128 + kg + i) * 128 + n];
  } else {
#pragma unroll
    for (int i = 0; i < 4; ++i) hv[i] = (_Float16)W2[(kg + i) * 128 + n];
  }
  *(v4h*)&w16[(size_t)g * 4] = hv;
}

// ---------------------------------------------------------------------------
// Fragment loader: ISA 16-bit layout. lane<16: K {0..7, 16..23} of row m=lane;
// lane>=16: K {8..15, 24..31}. Two 16B LDS reads per fragment.
// ---------------------------------------------------------------------------
__device__ __forceinline__ v16h load_frag(const _Float16* __restrict__ base) {
  union { v16h v; v8h h[2]; } u;
  u.h[0] = *(const v8h*)(base);
  u.h[1] = *(const v8h*)(base + 16);
  return u.v;
}

// ---------------------------------------------------------------------------
// Fused edge MLP: gather + 3-way GEMM (K=384) + SiLU + GEMM (K=128) + LN + res
// One 128-edge tile per workgroup (256 threads = 8 waves, wave32).
// ---------------------------------------------------------------------------
extern "C" __global__ __launch_bounds__(256)
void edge_mlp_wmma(const float* __restrict__ efeat, const float* __restrict__ nfeat,
                   const int* __restrict__ src_idx, const int* __restrict__ dst_idx,
                   const _Float16* __restrict__ w16,
                   const float* __restrict__ b1, const float* __restrict__ b2,
                   const float* __restrict__ ln_scale, const float* __restrict__ ln_bias,
                   float* __restrict__ out) {
  extern __shared__ char smem[];
  _Float16* sW = (_Float16*)(smem);            // 4 * 128 * 136 halfs (139264 B)
  _Float16* sA = (_Float16*)(smem + 139264);   // 128 * 136 halfs     (34816 B)
  _Float16* sH = (_Float16*)(smem + 174080);   // 128 * 136 halfs     (34816 B)
  float*    sY = (float*)   (smem + 208896);   // 128 * 132 floats    (67584 B)

  const int tid  = threadIdx.x;
  const int e0   = blockIdx.x * 128;

  // ---- stage f16 weights: 512 rows x 128 halfs -> LDS stride 136 ----
  for (int i = tid; i < 512 * 32; i += 256) {
    int r = i >> 5, kg = (i & 31) * 4;
    *(v4h*)&sW[r * LDSW + kg] = *(const v4h*)&w16[r * 128 + kg];
  }

  const int lane  = tid & 31;
  const int wave  = tid >> 5;
  const int wm    = (wave & 3) * 32;   // wave M base (4 waves along M)
  const int wn    = (wave >> 2) * 64;  // wave N base (2 waves along N)
  const int khalf = lane >> 4;         // K-half select per ISA layout
  const int l16   = lane & 15;

  v8f acc[2][4];
#pragma unroll
  for (int m = 0; m < 2; ++m)
#pragma unroll
    for (int n = 0; n < 4; ++n) acc[m][n] = (v8f){0,0,0,0,0,0,0,0};

  // ---- per-thread loader mapping: row = tid/2, 64 cols per thread ----
  const int lr = tid >> 1;
  const int lc = (tid & 1) * 64;
  int eL = e0 + lr; if (eL > EDGES - 1) eL = EDGES - 1;
  const int srow = src_idx[eL];
  const int drow = dst_idx[eL];
  const float* gsrc0 = efeat + (size_t)eL  * DIM;
  const float* gsrc1 = nfeat + (size_t)srow * DIM;
  const float* gsrc2 = nfeat + (size_t)drow * DIM;
  __builtin_prefetch(gsrc1 + lc, 0, 3);   // global_prefetch_b8
  __builtin_prefetch(gsrc2 + lc, 0, 3);

  // ---- GEMM1: h = efeat@Wf + src@Ws + dst@Wd (3 sources, K=128 each) ----
  for (int s = 0; s < 3; ++s) {
    const float* rp = (s == 0 ? gsrc0 : (s == 1 ? gsrc1 : gsrc2)) + lc;
#pragma unroll
    for (int j = 0; j < 16; ++j) {
      v4f f = *(const v4f*)(rp + j * 4);
      v4h hv = { (_Float16)f.x, (_Float16)f.y, (_Float16)f.z, (_Float16)f.w };
      *(v4h*)&sA[lr * LDSW + lc + j * 4] = hv;
    }
    __syncthreads();

    const _Float16* sB = sW + s * 128 * LDSW;  // transposed weights [n][k]
#pragma unroll
    for (int kb = 0; kb < 4; ++kb) {
      const int k0 = kb * 32 + khalf * 8;
      v16h af[2];
#pragma unroll
      for (int m = 0; m < 2; ++m)
        af[m] = load_frag(&sA[(wm + m * 16 + l16) * LDSW + k0]);
#pragma unroll
      for (int n = 0; n < 4; ++n) {
        v16h bf = load_frag(&sB[(wn + n * 16 + l16) * LDSW + k0]);
#pragma unroll
        for (int m = 0; m < 2; ++m)
          acc[m][n] = __builtin_amdgcn_wmma_f32_16x16x32_f16(
              false, af[m], false, bf, (short)0, acc[m][n], false, false);
      }
    }
    __syncthreads();   // protect sA before next source overwrites it
  }

  // ---- bias + SiLU -> sH (f16) ----
  float b1v[4];
#pragma unroll
  for (int n = 0; n < 4; ++n) b1v[n] = b1[wn + n * 16 + l16];
#pragma unroll
  for (int m = 0; m < 2; ++m)
#pragma unroll
    for (int n = 0; n < 4; ++n)
#pragma unroll
      for (int g = 0; g < 8; ++g) {
        float x = acc[m][n][g] + b1v[n];
        float sl = x / (1.0f + __expf(-x));   // silu
        sH[(wm + m * 16 + g + khalf * 8) * LDSW + wn + n * 16 + l16] = (_Float16)sl;
      }
  __syncthreads();

  // ---- GEMM2: y = h @ W2 (K=128) ----
  v8f acc2[2][4];
#pragma unroll
  for (int m = 0; m < 2; ++m)
#pragma unroll
    for (int n = 0; n < 4; ++n) acc2[m][n] = (v8f){0,0,0,0,0,0,0,0};

  const _Float16* sB2 = sW + 3 * 128 * LDSW;
#pragma unroll
  for (int kb = 0; kb < 4; ++kb) {
    const int k0 = kb * 32 + khalf * 8;
    v16h af[2];
#pragma unroll
    for (int m = 0; m < 2; ++m)
      af[m] = load_frag(&sH[(wm + m * 16 + l16) * LDSW + k0]);
#pragma unroll
    for (int n = 0; n < 4; ++n) {
      v16h bf = load_frag(&sB2[(wn + n * 16 + l16) * LDSW + k0]);
#pragma unroll
      for (int m = 0; m < 2; ++m)
        acc2[m][n] = __builtin_amdgcn_wmma_f32_16x16x32_f16(
            false, af[m], false, bf, (short)0, acc2[m][n], false, false);
    }
  }

  // ---- + b2 -> sY (f32) for row-wise LayerNorm ----
  float b2v[4];
#pragma unroll
  for (int n = 0; n < 4; ++n) b2v[n] = b2[wn + n * 16 + l16];
#pragma unroll
  for (int m = 0; m < 2; ++m)
#pragma unroll
    for (int n = 0; n < 4; ++n)
#pragma unroll
      for (int g = 0; g < 8; ++g)
        sY[(wm + m * 16 + g + khalf * 8) * YSTR + wn + n * 16 + l16] =
            acc2[m][n][g] + b2v[n];
  __syncthreads();

  // ---- LayerNorm + residual + store (thread pair per row, shfl combine) ----
  {
    const int r  = tid >> 1;
    const int cb = (tid & 1) * 64;
    float sum = 0.0f, sq = 0.0f;
#pragma unroll
    for (int j = 0; j < 16; ++j) {
      v4f y = *(const v4f*)&sY[r * YSTR + cb + j * 4];
      sum += y.x + y.y + y.z + y.w;
      sq  += y.x * y.x + y.y * y.y + y.z * y.z + y.w * y.w;
    }
    sum += __shfl_xor(sum, 1, 32);
    sq  += __shfl_xor(sq, 1, 32);
    const float mu   = sum * (1.0f / 128.0f);
    const float var  = sq * (1.0f / 128.0f) - mu * mu;
    const float rstd = rsqrtf(var + 1e-5f);

    const int e = e0 + r;
    if (e < EDGES) {
      const float* er   = efeat + (size_t)e * DIM + cb;
      float*       orow = out   + (size_t)e * DIM + cb;
#pragma unroll
      for (int j = 0; j < 16; ++j) {
        v4f y  = *(const v4f*)&sY[r * YSTR + cb + j * 4];
        v4f sc = *(const v4f*)(ln_scale + cb + j * 4);
        v4f bi = *(const v4f*)(ln_bias  + cb + j * 4);
        v4f ef = *(const v4f*)(er + j * 4);
        v4f o;
        o.x = (y.x - mu) * rstd * sc.x + bi.x + ef.x;
        o.y = (y.y - mu) * rstd * sc.y + bi.y + ef.y;
        o.z = (y.z - mu) * rstd * sc.z + bi.z + ef.z;
        o.w = (y.w - mu) * rstd * sc.w + bi.w + ef.w;
        *(v4f*)(orow + j * 4) = o;
      }
    }
  }
}

extern "C" void kernel_launch(void* const* d_in, const int* in_sizes, int n_in,
                              void* d_out, int out_size, void* d_ws, size_t ws_size,
                              hipStream_t stream) {
  const float* efeat    = (const float*)d_in[0];
  const float* nfeat    = (const float*)d_in[1];
  const int*   src_idx  = (const int*)d_in[2];
  const int*   dst_idx  = (const int*)d_in[3];
  const float* W1       = (const float*)d_in[4];
  const float* b1       = (const float*)d_in[5];
  const float* W2       = (const float*)d_in[6];
  const float* b2       = (const float*)d_in[7];
  const float* ln_scale = (const float*)d_in[8];
  const float* ln_bias  = (const float*)d_in[9];
  float* out = (float*)d_out;
  _Float16* w16 = (_Float16*)d_ws;   // needs 4*128*128*2 = 128 KB scratch

  // 1) convert + transpose weights to f16 once
  prep_weights_f16<<<64, 256, 0, stream>>>(W1, W2, w16);

  // 2) fused edge MLP, one 128-edge tile per workgroup
  const int nblocks = (EDGES + 127) / 128;          // 4688
  const size_t smem = 276480;                       // 270 KB of the 320 KB WGP LDS
  edge_mlp_wmma<<<nblocks, 256, smem, stream>>>(efeat, nfeat, src_idx, dst_idx, w16,
                                                b1, b2, ln_scale, ln_bias, out);

  // 3) nfeat passthrough (second tuple output)
  hipMemcpyAsync(out + (size_t)EDGES * DIM, nfeat,
                 (size_t)NNODES * DIM * sizeof(float),
                 hipMemcpyDeviceToDevice, stream);
}
